// Grapher_88725434401297
// MI455X (gfx1250) — matrix operand order repaired and checked by reading, not compiled, not run on hardware
//
#include <hip/hip_runtime.h>
#include <hip/hip_bf16.h>

typedef __attribute__((ext_vector_type(16))) _Float16 v16h;
typedef __attribute__((ext_vector_type(8)))  float    v8f;

#define N_NODES 65536
#define N_EDGES 1048576
#define E_TOT   (N_EDGES + N_NODES)   // 1114112 (self loops appended)
#define VOCAB   50000
#define HID     256
#define G_CNT   64
#define NEG_SLOPE 0.2f
#define EPS_SM  1e-16f

// ---------- ordered-float encoding for atomic segment-max ----------
__device__ __forceinline__ unsigned enc_f(float f) {
  unsigned u = __float_as_uint(f);
  return (u & 0x80000000u) ? ~u : (u | 0x80000000u);
}
__device__ __forceinline__ float dec_f(unsigned u) {
  unsigned b = (u & 0x80000000u) ? (u & 0x7FFFFFFFu) : ~u;
  return __uint_as_float(b);
}
#define ENC_NEG_INF 0x007FFFFFu  // enc(-inf)

// ---------- generic fill ----------
__global__ void fill_u32_k(unsigned* p, unsigned v, size_t n) {
  size_t i = (size_t)blockIdx.x * blockDim.x + threadIdx.x;
  size_t st = (size_t)gridDim.x * blockDim.x;
  for (; i < n; i += st) p[i] = v;
}

// ---------- embedding gather ----------
__global__ void gather_emb_k(const int* __restrict__ nodes,
                             const float* __restrict__ emb,
                             float* __restrict__ x) {
  int n = blockIdx.x, f = threadIdx.x;
  x[(size_t)n * HID + f] = emb[(size_t)nodes[n] * HID + f];
}

// ---------- WMMA fragment helpers (wave32; A/B share contiguous k = 16*half + s order) ----------
// A fragment read from LDS-staged f16 tile [16][256].
// lanes0-15: M=lane,  VGPR0-7 hold K=kBase+0..15
// lanes16-31: M=lane-16, K=kBase+16..31.  Each 8-half group is one b128 LDS read.
__device__ __forceinline__ v16h lds_a_frag(const _Float16* __restrict__ As, int kBase) {
  int lane = threadIdx.x & 31;
  int half_id = lane >> 4;
  int m = lane & 15;
  const _Float16* p = As + m * HID + kBase + half_id * 16;
  v16h a;
#pragma unroll
  for (int j = 0; j < 8; ++j) a[j] = p[j];
#pragma unroll
  for (int j = 0; j < 8; ++j) a[8 + j] = p[8 + j];
  return a;
}
// B: 32x16 f16 (K x N). lanes0-15: K=0..15 (2/VGPR), N=lane; lanes16-31: K=16..31.
// LDB is compile-time: row strides become 24-bit instruction immediate offsets.
template <int LDB>
__device__ __forceinline__ v16h load_b_frag(const float* __restrict__ B, int kBase, int nBase) {
  int lane = threadIdx.x & 31;
  int half_id = lane >> 4;
  const float* p = B + (size_t)(kBase + half_id * 16) * LDB + nBase + (lane & 15);
  v16h b;
#pragma unroll
  for (int v = 0; v < 16; ++v)
    b[v] = (_Float16)p[(size_t)v * LDB];     // constant byte offsets (<= 6.2 MB < 2^23)
  return b;
}
// C/D f32: VGPR r -> M = half*8 + r, N = lane&15
template <bool BIASO, int LDC>
__device__ __forceinline__ void store_c_frag(float* __restrict__ C, int nBase,
                                             v8f c, const float* __restrict__ biasOut) {
  int lane = threadIdx.x & 31;
  int half_id = lane >> 4;
  int n = (lane & 15) + nBase;
  float bo = BIASO ? biasOut[n] : 0.f;
  float* p = C + (size_t)(half_id * 8) * LDC + n;
#pragma unroll
  for (int r = 0; r < 8; ++r)
    p[(size_t)r * LDC] = c[r] + bo;          // constant byte offsets
}

// ---------- WMMA GEMM: C[M x N] = op(A)[M x 256] @ B[256 x N] (+biasOut) ----------
// op(A) = RELUA ? relu(A + biasA) : A, fused into the one-time LDS staging pass.
// block = 128 thr (4 waves). Block owns one 16-row tile; A staged once in LDS as
// f16 (8 KB). Each wave computes 4 16x16 column tiles as two dual-accumulator
// pairs (shared A fragment, interleaved WMMAs to stretch the D->C RAW distance).
// All tile guards are wave-uniform so EXEC stays all-ones for WMMA.
// grid.x = ceil(Ntiles/16), grid.y = M/16.
template <bool BIASA, bool RELUA, bool BIASO, int N>
__global__ void gemm_wmma_k(const float* __restrict__ A, const float* __restrict__ biasA,
                            const float* __restrict__ B, const float* __restrict__ biasOut,
                            float* __restrict__ C) {
  __shared__ _Float16 As[16 * HID];  // 8 KB
  const int rowBase = blockIdx.y * 16;
  const float* Arow = A + (size_t)rowBase * HID;

  // cooperative stage: 16x256 f32 -> f16 (+bias+relu), 2 elems/thread/iter
#pragma unroll
  for (int i = threadIdx.x; i < 16 * (HID / 2); i += 128) {
    int m = i >> 7;               // / (HID/2)
    int k = (i & 127) * 2;
    float2 f = *(const float2*)(Arow + (size_t)m * HID + k);
    float f0 = f.x, f1 = f.y;
    if (BIASA) { float2 bb = *(const float2*)(biasA + k); f0 += bb.x; f1 += bb.y; }
    if (RELUA) { f0 = fmaxf(f0, 0.f); f1 = fmaxf(f1, 0.f); }
    As[m * HID + k]     = (_Float16)f0;
    As[m * HID + k + 1] = (_Float16)f1;
  }
  __syncthreads();

  float* Crow = C + (size_t)rowBase * N;
  const int wave = threadIdx.x >> 5;
#pragma unroll
  for (int tp = 0; tp < 2; ++tp) {
    const int ct0 = blockIdx.x * 16 + wave * 4 + tp * 2;
    const int ct1 = ct0 + 1;
    const bool ok0 = ct0 * 16 < N;     // wave-uniform
    const bool ok1 = ct1 * 16 < N;
    if (!ok0) continue;
    if (ok1) {
      v8f c0 = {}, c1 = {};
#pragma unroll
      for (int kt = 0; kt < HID; kt += 32) {
        v16h a  = lds_a_frag(As, kt);
        v16h b0 = load_b_frag<N>(B, kt, ct0 * 16);
        v16h b1 = load_b_frag<N>(B, kt, ct1 * 16);
        c0 = __builtin_amdgcn_wmma_f32_16x16x32_f16(false, a, false, b0, (short)0, c0,
                                                    false, false);
        c1 = __builtin_amdgcn_wmma_f32_16x16x32_f16(false, a, false, b1, (short)0, c1,
                                                    false, false);
      }
      store_c_frag<BIASO, N>(Crow, ct0 * 16, c0, biasOut);
      store_c_frag<BIASO, N>(Crow, ct1 * 16, c1, biasOut);
    } else {
      v8f c0 = {};
#pragma unroll
      for (int kt = 0; kt < HID; kt += 32) {
        v16h a  = lds_a_frag(As, kt);
        v16h b0 = load_b_frag<N>(B, kt, ct0 * 16);
        c0 = __builtin_amdgcn_wmma_f32_16x16x32_f16(false, a, false, b0, (short)0, c0,
                                                    false, false);
      }
      store_c_frag<BIASO, N>(Crow, ct0 * 16, c0, biasOut);
    }
  }
}

// ---------- per-node attention dots: es = h.a_s, ed = h.a_d ----------
__global__ void node_dots_k(const float* __restrict__ h, const float* __restrict__ a_s,
                            const float* __restrict__ a_d, float* __restrict__ es,
                            float* __restrict__ ed) {
  int wave = threadIdx.x >> 5, lane = threadIdx.x & 31;
  int n = blockIdx.x * (blockDim.x >> 5) + wave;
  if (n >= N_NODES) return;
  const float* hp = h + (size_t)n * HID;
  float s0 = 0.f, s1 = 0.f;
#pragma unroll
  for (int f = lane; f < HID; f += 32) {
    float v = hp[f];
    s0 += v * a_s[f];
    s1 += v * a_d[f];
  }
#pragma unroll
  for (int o = 16; o; o >>= 1) {
    s0 += __shfl_xor(s0, o, 32);
    s1 += __shfl_xor(s1, o, 32);
  }
  if (lane == 0) { es[n] = s0; ed[n] = s1; }
}

// ---------- edge pass 1: e = leaky_relu(es[src]+ed[dst]); segment max ----------
__global__ void edge_score_max_k(const int* __restrict__ src, const int* __restrict__ dst,
                                 const float* __restrict__ es, const float* __restrict__ ed,
                                 float* __restrict__ ebuf, unsigned* __restrict__ encmax) {
  int e = blockIdx.x * blockDim.x + threadIdx.x;
  if (e >= E_TOT) return;
  int s, d;
  if (e < N_EDGES) { s = src[e]; d = dst[e]; } else { s = d = e - N_EDGES; }
  float v = es[s] + ed[d];
  v = (v > 0.f) ? v : NEG_SLOPE * v;
  ebuf[e] = v;
  atomicMax(&encmax[d], enc_f(v));
}

// ---------- edge pass 2: w = exp(e - max[dst]); segment sum ----------
__global__ void edge_exp_sum_k(const int* __restrict__ dst, float* __restrict__ ebuf,
                               const unsigned* __restrict__ encmax, float* __restrict__ denom) {
  int e = blockIdx.x * blockDim.x + threadIdx.x;
  if (e >= E_TOT) return;
  int d = (e < N_EDGES) ? dst[e] : e - N_EDGES;
  float w = expf(ebuf[e] - dec_f(encmax[d]));
  ebuf[e] = w;
  atomicAdd(&denom[d], w);
}

// ---------- edge pass 3: agg[dst] += h[src] * alpha (one wave per edge) ----------
__global__ void edge_aggregate_k(const int* __restrict__ src, const int* __restrict__ dst,
                                 const float* __restrict__ h, const float* __restrict__ ebuf,
                                 const float* __restrict__ denom, float* __restrict__ agg) {
  int wave = threadIdx.x >> 5, lane = threadIdx.x & 31;
  int e = blockIdx.x * (blockDim.x >> 5) + wave;
  if (e >= E_TOT) return;
  int s, d;
  if (e < N_EDGES) { s = src[e]; d = dst[e]; } else { s = d = e - N_EDGES; }
  float alpha = ebuf[e] / (denom[d] + EPS_SM);
  const float* hs = h + (size_t)s * HID;
  float* ad = agg + (size_t)d * HID;
#pragma unroll
  for (int f = lane; f < HID; f += 32)
    atomicAdd(&ad[f], hs[f] * alpha);
}

// ---------- mean pool ----------
__global__ void pool_sum_k(const float* __restrict__ x, const int* __restrict__ batch,
                           float* __restrict__ pooled, float* __restrict__ cnt) {
  int wave = threadIdx.x >> 5, lane = threadIdx.x & 31;
  int n = blockIdx.x * (blockDim.x >> 5) + wave;
  if (n >= N_NODES) return;
  int g = batch[n];
  const float* xp = x + (size_t)n * HID;
  float* pp = pooled + (size_t)g * HID;
#pragma unroll
  for (int f = lane; f < HID; f += 32)
    atomicAdd(&pp[f], xp[f]);
  if (lane == 0) atomicAdd(&cnt[g], 1.0f);
}
__global__ void pool_div_k(float* __restrict__ pooled, const float* __restrict__ cnt) {
  int g = blockIdx.x, f = threadIdx.x;
  pooled[(size_t)g * HID + f] /= fmaxf(cnt[g], 1.0f);
}

extern "C" void kernel_launch(void* const* d_in, const int* in_sizes, int n_in,
                              void* d_out, int out_size, void* d_ws, size_t ws_size,
                              hipStream_t stream) {
  (void)in_sizes; (void)n_in; (void)out_size; (void)ws_size;
  const int*   nodes  = (const int*)  d_in[0];
  const int*   eidx   = (const int*)  d_in[1];   // [2][N_EDGES]: row0=src, row1=dst
  const int*   batch  = (const int*)  d_in[2];
  const float* emb    = (const float*)d_in[3];
  const float* Wg     = (const float*)d_in[4];   // [3][256][256]
  const float* a_src  = (const float*)d_in[5];   // [3][256]
  const float* a_dst  = (const float*)d_in[6];
  const float* bg     = (const float*)d_in[7];
  const float* Wl     = (const float*)d_in[8];
  const float* bl     = (const float*)d_in[9];
  const float* Wout   = (const float*)d_in[10];  // [256][50000]
  const float* bout   = (const float*)d_in[11];
  float* out = (float*)d_out;                    // [64][50000]

  const int* srcv = eidx;
  const int* dstv = eidx + N_EDGES;

  // workspace carve-up (floats)
  const size_t NH = (size_t)N_NODES * HID;       // 16,777,216
  float*    bufA   = (float*)d_ws;               // x / new x
  float*    bufB   = bufA + NH;                  // h
  float*    bufC   = bufB + NH;                  // agg
  float*    es     = bufC + NH;                  // [N]
  float*    ed     = es + N_NODES;               // [N]
  unsigned* encmax = (unsigned*)(ed + N_NODES);  // [N]
  float*    denom  = (float*)(encmax + N_NODES); // [N]
  float*    ebuf   = denom + N_NODES;            // [E_TOT]
  float*    pooled = ebuf + E_TOT;               // [64][256]
  float*    cnt    = pooled + (size_t)G_CNT * HID; // [64]

  // 1) x = emb[nodes]
  gather_emb_k<<<N_NODES, HID, 0, stream>>>(nodes, emb, bufA);

  dim3 gemmBlk(128);
  dim3 gemmGridNode(1, N_NODES / 16);            // 16 col tiles per block, 4096 row tiles

  for (int l = 0; l < 3; ++l) {
    const float* Wgl = Wg + (size_t)l * HID * HID;
    const float* Wll = Wl + (size_t)l * HID * HID;
    const float* bgl = bg + (size_t)l * HID;
    const float* bll = bl + (size_t)l * HID;
    const float* asl = a_src + (size_t)l * HID;
    const float* adl = a_dst + (size_t)l * HID;

    // h = x @ Wg[l]
    gemm_wmma_k<false, false, false, HID>
        <<<gemmGridNode, gemmBlk, 0, stream>>>(bufA, nullptr, Wgl, nullptr, bufB);
    // per-node attention scalars
    node_dots_k<<<N_NODES / 8, 256, 0, stream>>>(bufB, asl, adl, es, ed);
    // init segment buffers + agg
    fill_u32_k<<<1024, 256, 0, stream>>>(encmax, ENC_NEG_INF, N_NODES);
    fill_u32_k<<<1024, 256, 0, stream>>>((unsigned*)denom, 0u, N_NODES);
    fill_u32_k<<<4096, 256, 0, stream>>>((unsigned*)bufC, 0u, NH);
    // edge phases
    edge_score_max_k<<<E_TOT / 256, 256, 0, stream>>>(srcv, dstv, es, ed, ebuf, encmax);
    edge_exp_sum_k<<<E_TOT / 256, 256, 0, stream>>>(dstv, ebuf, encmax, denom);
    edge_aggregate_k<<<E_TOT / 8, 256, 0, stream>>>(srcv, dstv, bufB, ebuf, denom, bufC);
    // x = relu(agg + bg) @ Wl + bl   (bias+relu fused into LDS staging, f32)
    gemm_wmma_k<true, true, true, HID>
        <<<gemmGridNode, gemmBlk, 0, stream>>>(bufC, bgl, Wll, bll, bufA);
  }

  // mean pool
  fill_u32_k<<<64, 256, 0, stream>>>((unsigned*)pooled, 0u, (size_t)G_CNT * HID);
  fill_u32_k<<<1, 64, 0, stream>>>((unsigned*)cnt, 0u, G_CNT);
  pool_sum_k<<<N_NODES / 8, 256, 0, stream>>>(bufA, batch, pooled, cnt);
  pool_div_k<<<G_CNT, HID, 0, stream>>>(pooled, cnt);

  // logits = pooled @ Wout + bout : 50000 = 3125 column tiles, 16 per block
  dim3 gemmGridOut((VOCAB / 16 + 15) / 16, G_CNT / 16);  // (196, 4), tail guarded
  gemm_wmma_k<false, false, true, VOCAB>
      <<<gemmGridOut, gemmBlk, 0, stream>>>(pooled, nullptr, Wout, bout, out);
}
